// Skin_Non_Local_10015863734431
// MI455X (gfx1250) — matrix operand order repaired and checked by reading, not compiled, not run on hardware
//
#include <hip/hip_runtime.h>
#include <hip/hip_bf16.h>

#define B_  4
#define C_  256
#define C2_ 128
#define N_  4096

typedef __attribute__((ext_vector_type(16))) _Float16 v16h;
typedef __attribute__((ext_vector_type(8)))  _Float16 v8h;
typedef __attribute__((ext_vector_type(8)))  float    v8f;
typedef int v4i_ __attribute__((vector_size(16)));   // matches builtin's int4 pointee

// ---- async global->LDS staging (gfx1250 path, guarded) ----------------------
#if __has_builtin(__builtin_amdgcn_global_load_async_to_lds_b128)
#define HAVE_ASYNC 1
#define ASYNC_COPY16(dst, src)                                                  \
  __builtin_amdgcn_global_load_async_to_lds_b128(                               \
      (__attribute__((address_space(1))) v4i_*)(src),                           \
      (__attribute__((address_space(3))) v4i_*)(dst), 0, 0)
#else
#define HAVE_ASYNC 0
#define ASYNC_COPY16(dst, src) (*(v8h*)(dst) = *(const v8h*)(src))
#endif

#if HAVE_ASYNC
#if __has_builtin(__builtin_amdgcn_s_wait_asynccnt)
#define WAIT_ASYNC(n) __builtin_amdgcn_s_wait_asynccnt(n)
#else
#define WAIT_ASYNC(n) asm volatile("s_wait_asynccnt %0" ::"i"(n) : "memory")
#endif
#else
#define WAIT_ASYNC(n)
#endif

__device__ __forceinline__ v8f wmma16(v16h a, v16h b, v8f c) {
  // D = A(16x32 f16) * B(32x16 f16) + C(16x16 f32)
  return __builtin_amdgcn_wmma_f32_16x16x32_f16(false, a, false, b, (short)0, c, false, false);
}

// ---- fragment loaders -------------------------------------------------------
// A-matrix (16x32, f16) per ISA layout: lane<16 -> row=lane&15, K=0..7,16..23
//                                       lane>=16 -> same row,  K=8..15,24..31
__device__ __forceinline__ v16h load_afrag16(const _Float16* p, int stride, int rowbase, int colbase) {
  const int lane = threadIdx.x & 31;
  const int row  = rowbase + (lane & 15);
  const int koff = (lane < 16) ? 0 : 8;
  const _Float16* q = p + (size_t)row * stride + colbase + koff;
  v8h lo = *(const v8h*)(q);
  v8h hi = *(const v8h*)(q + 16);
  v16h r;
#pragma unroll
  for (int i = 0; i < 8; ++i) { r[i] = lo[i]; r[8 + i] = hi[i]; }
  return r;
}

__device__ __forceinline__ v16h load_afrag32(const float* p, int stride, int rowbase, int colbase) {
  const int lane = threadIdx.x & 31;
  const int row  = rowbase + (lane & 15);
  const int koff = (lane < 16) ? 0 : 8;
  const float* q = p + (size_t)row * stride + colbase + koff;
  v16h r;
#pragma unroll
  for (int i = 0; i < 8; ++i) { r[i] = (_Float16)q[i]; r[8 + i] = (_Float16)q[16 + i]; }
  return r;
}

// B-matrix (32x16, f16): source stored as [col][k] row-major.
// lanes 0-15: col=lane, K=0..7,16..23 ; lanes 16-31: col=lane-16, K=8..15,24..31
__device__ __forceinline__ v16h load_bfrag16(const _Float16* p, int stride, int colbase, int kbase) {
  const int lane = threadIdx.x & 31;
  const int col  = colbase + (lane & 15);
  const int k0   = kbase + ((lane < 16) ? 0 : 8);
  const _Float16* q = p + (size_t)col * stride + k0;
  v8h lo = *(const v8h*)(q);
  v8h hi = *(const v8h*)(q + 16);
  v16h r;
#pragma unroll
  for (int i = 0; i < 8; ++i) { r[i] = lo[i]; r[8 + i] = hi[i]; }
  return r;
}

__device__ __forceinline__ v16h load_bfrag32(const float* p, int stride, int colbase, int kbase) {
  const int lane = threadIdx.x & 31;
  const int col  = colbase + (lane & 15);
  const float* q = p + (size_t)col * stride + kbase + ((lane < 16) ? 0 : 8);
  v16h r;
#pragma unroll
  for (int i = 0; i < 16; ++i) r[i] = (_Float16)q[i < 8 ? i : i + 8];
  return r;
}

// ---- kernel 1: 1x1-conv projections -> f16 Q[n][c2], K[n][c2] (masked), V[c][n]
__global__ __launch_bounds__(256) void proj_kernel(
    const float* __restrict__ x, const float* __restrict__ mask, const float* __restrict__ skin,
    const float* __restrict__ Wq, const float* __restrict__ bq,
    const float* __restrict__ Wk, const float* __restrict__ bk,
    const float* __restrict__ Wv, const float* __restrict__ bv,
    _Float16* __restrict__ Qh, _Float16* __restrict__ Kh, _Float16* __restrict__ Vh)
{
  __shared__ _Float16 xT[64 * C_]; // 32 KB: x tile transposed [n][c], f16
  const int b  = blockIdx.x;
  const int n0 = blockIdx.y * 64;
  const int tid = threadIdx.x;

  { // stage + convert + transpose
    const int n = tid & 63;
    for (int c = tid >> 6; c < C_; c += 4)
      xT[n * C_ + c] = (_Float16)x[((size_t)b * C_ + c) * N_ + n0 + n];
  }
  __syncthreads();

  const int wave = tid >> 5, lane = tid & 31;
  const int half8 = (lane < 16) ? 0 : 8;

  for (int job = wave; job < 128; job += 8) {
    if (job < 64) {            // ---- Q (jobs 0..31) or K (jobs 32..63): D[o][n]
      const bool isK = job >= 32;
      const int j  = job & 31;
      const int ot = j & 7, nt = j >> 3;
      const int obase = ot * 16, ntb = nt * 16;
      const float* W    = isK ? Wk : Wq;
      const float* bias = isK ? bk : bq;
      v8f acc;
#pragma unroll
      for (int r = 0; r < 8; ++r) acc[r] = bias[obase + half8 + r];
#pragma unroll
      for (int kb = 0; kb < 8; ++kb) {
        v16h a  = load_afrag32(W,  C_, obase, kb * 32); // A = W rows (o x c)
        v16h bf = load_bfrag16(xT, C_, ntb,   kb * 32); // B = x (c x n)
        acc = wmma16(a, bf, acc);
      }
      const int n = n0 + ntb + (lane & 15);
      float fac = 1.0f;
      if (isK) fac = (1.0f - mask[(size_t)b * N_ + n]) * skin[(size_t)b * N_ + n];
      v8h ov;
#pragma unroll
      for (int r = 0; r < 8; ++r) ov[r] = (_Float16)(acc[r] * fac);
      _Float16* dst = (isK ? Kh : Qh) + ((size_t)b * N_ + n) * C2_ + obase + half8;
      *(v8h*)dst = ov;
    } else {                   // ---- V (jobs 64..127): D[n][o]
      const int j  = job - 64;
      const int nt = j & 3, ot = j >> 2;
      const int ntb = nt * 16, obase = ot * 16;
      const int o = obase + (lane & 15);
      v8f acc;
      const float bval = bv[o];
#pragma unroll
      for (int r = 0; r < 8; ++r) acc[r] = bval;
#pragma unroll
      for (int kb = 0; kb < 8; ++kb) {
        v16h a  = load_afrag16(xT, C_, ntb,   kb * 32); // A = x^T (n x c)
        v16h bf = load_bfrag32(Wv, C_, obase, kb * 32); // B = Wv^T (c x o)
        acc = wmma16(a, bf, acc);
      }
      v8h ov;
#pragma unroll
      for (int r = 0; r < 8; ++r) ov[r] = (_Float16)acc[r];
      *(v8h*)(Vh + ((size_t)b * C_ + o) * N_ + n0 + ntb + half8) = ov;
    }
  }
}

// ---- kernel 2: fused flash attention + epilogue -----------------------------
// wave = 16 queries x all 256 channels; 8 waves/WG -> 128 queries/WG.
// K/V tiles staged cooperatively in LDS via async DMA, double-buffered.
__global__ __launch_bounds__(256) void attn_kernel(
    const _Float16* __restrict__ Qh, const _Float16* __restrict__ Kh, const _Float16* __restrict__ Vh,
    const float* __restrict__ x, const float* __restrict__ mask, const float* __restrict__ skin,
    const float* __restrict__ gamma, float* __restrict__ out)
{
  __shared__ _Float16 kbuf[2][32 * C2_]; // 8 KB / buffer: K tile [key][c]
  __shared__ _Float16 vbuf[2][C_ * 32];  // 16 KB / buffer: V tile [c][key]
  __shared__ _Float16 pbuf[8][16 * 32];  // per-wave P staging (8 KB)

  const int b = blockIdx.x;
  const int tid = threadIdx.x;
  const int wave = tid >> 5, lane = tid & 31;
  const int half8 = (lane < 16) ? 0 : 8;
  const int qbase = blockIdx.y * 128 + wave * 16;

  const _Float16* Qb = Qh + (size_t)b * N_ * C2_;
  const _Float16* Kb = Kh + (size_t)b * N_ * C2_;
  const _Float16* Vb = Vh + (size_t)b * C_ * N_;

  // cooperative stage of one 32-key tile: 6 async B128 per thread
  auto stage = [&](int d, int kk) {
#pragma unroll
    for (int t = 0; t < 2; ++t) {           // K: 512 x 16B chunks
      const int ci = tid + t * 256;
      const int key = ci >> 4, coff = (ci & 15) * 8;
      ASYNC_COPY16(&kbuf[d][key * C2_ + coff], Kb + (size_t)(kk + key) * C2_ + coff);
    }
#pragma unroll
    for (int t = 0; t < 4; ++t) {           // V: 1024 x 16B chunks
      const int ci = tid + t * 256;
      const int c = ci >> 2, koff = (ci & 3) * 8;
      ASYNC_COPY16(&vbuf[d][c * 32 + koff], Vb + (size_t)c * N_ + kk + koff);
    }
  };

  v16h qa[4];
#pragma unroll
  for (int kb = 0; kb < 4; ++kb) qa[kb] = load_afrag16(Qb, C2_, qbase, kb * 32);

  v8f O[16];
#pragma unroll
  for (int t = 0; t < 16; ++t)
#pragma unroll
    for (int i = 0; i < 8; ++i) O[t][i] = 0.0f;

  float rm[8], l[8];
#pragma unroll
  for (int r = 0; r < 8; ++r) { rm[r] = -1.0e30f; l[r] = 0.0f; }

  _Float16* pw = &pbuf[wave][0];

  stage(0, 0);
  for (int it = 0; it < N_ / 32; ++it) {
    const int d = it & 1;
    if (it + 1 < N_ / 32) {
      stage(d ^ 1, (it + 1) * 32);  // overlap next tile's DMA with compute
      WAIT_ASYNC(6);                // this wave's batch `it` retired (in-order)
    } else {
      WAIT_ASYNC(0);
    }
    __syncthreads();                // every wave's batch `it` is in LDS

    const _Float16* kt = &kbuf[d][0];
    const _Float16* vt = &vbuf[d][0];

    // S = Q . K^T  (two 16x16 tiles covering these 32 keys)
    v8f s0, s1;
#pragma unroll
    for (int i = 0; i < 8; ++i) { s0[i] = 0.0f; s1[i] = 0.0f; }
#pragma unroll
    for (int kb = 0; kb < 4; ++kb) {
      v16h k0 = load_bfrag16(kt, C2_, 0,  kb * 32);
      v16h k1 = load_bfrag16(kt, C2_, 16, kb * 32);
      s0 = wmma16(qa[kb], k0, s0);
      s1 = wmma16(qa[kb], k1, s1);
    }
    // streaming softmax over this 32-key block (row stats across 16 lanes)
    float sc[8];
#pragma unroll
    for (int r = 0; r < 8; ++r) {
      float t = fmaxf(s0[r], s1[r]);
      t = fmaxf(t, __shfl_xor(t, 1, 16));
      t = fmaxf(t, __shfl_xor(t, 2, 16));
      t = fmaxf(t, __shfl_xor(t, 4, 16));
      t = fmaxf(t, __shfl_xor(t, 8, 16));
      const float nm  = fmaxf(rm[r], t);
      const float scr = __expf(rm[r] - nm);
      rm[r] = nm;
      const float p0 = __expf(s0[r] - nm);
      const float p1 = __expf(s1[r] - nm);
      float ts = p0 + p1;
      ts += __shfl_xor(ts, 1, 16);
      ts += __shfl_xor(ts, 2, 16);
      ts += __shfl_xor(ts, 4, 16);
      ts += __shfl_xor(ts, 8, 16);
      l[r] = l[r] * scr + ts;
      sc[r] = scr;
      const int row = r + half8, col = lane & 15;
      pw[row * 32 + col]      = (_Float16)p0;
      pw[row * 32 + 16 + col] = (_Float16)p1;
    }
    asm volatile("s_wait_dscnt 0" ::: "memory"); // wave-local LDS RAW on P
    // reload P in A-fragment layout (16 queries x 32 keys)
    v16h pf;
    {
      const int m = lane & 15;
      const int koff = (lane < 16) ? 0 : 8;
      const _Float16* q = pw + m * 32 + koff;
      v8h lo = *(const v8h*)q;
      v8h hi = *(const v8h*)(q + 16);
#pragma unroll
      for (int i = 0; i < 8; ++i) { pf[i] = lo[i]; pf[8 + i] = hi[i]; }
    }
    // O[ct] = O[ct]*scale + P . V^T  over 16 channel tiles
#pragma unroll
    for (int ct = 0; ct < 16; ++ct) {
      v16h vf = load_bfrag16(vt, 32, ct * 16, 0);
      v8f o = O[ct];
#pragma unroll
      for (int r = 0; r < 8; ++r) o[r] *= sc[r];
      O[ct] = wmma16(pf, vf, o);
    }
    __syncthreads();                // buffer d is restaged next iteration
  }

  // epilogue: out = gamma * (O/l) * mask * skin + x
  const float g = gamma[0];
  float linv[8], mfac[8];
#pragma unroll
  for (int r = 0; r < 8; ++r) {
    const int m = qbase + half8 + r;
    mfac[r] = mask[(size_t)b * N_ + m] * skin[(size_t)b * N_ + m];
    linv[r] = 1.0f / l[r];
  }
#pragma unroll
  for (int ct = 0; ct < 16; ++ct) {
    const int c = ct * 16 + (lane & 15);
    const float* xp = x   + ((size_t)b * C_ + c) * N_ + qbase + half8;
    float*       op = out + ((size_t)b * C_ + c) * N_ + qbase + half8;
#pragma unroll
    for (int r = 0; r < 8; ++r)
      op[r] = g * (O[ct][r] * linv[r]) * mfac[r] + xp[r];
  }
}

extern "C" void kernel_launch(void* const* d_in, const int* in_sizes, int n_in,
                              void* d_out, int out_size, void* d_ws, size_t ws_size,
                              hipStream_t stream) {
  (void)in_sizes; (void)n_in; (void)out_size; (void)ws_size;
  const float* x    = (const float*)d_in[0];
  const float* mask = (const float*)d_in[1];
  const float* skin = (const float*)d_in[2];
  const float* Wq   = (const float*)d_in[3];
  const float* bq   = (const float*)d_in[4];
  const float* Wk   = (const float*)d_in[5];
  const float* bk   = (const float*)d_in[6];
  const float* Wv   = (const float*)d_in[7];
  const float* bv   = (const float*)d_in[8];
  const float* gamma= (const float*)d_in[9];
  float* out = (float*)d_out;

  char* ws = (char*)d_ws;
  const size_t qkBytes = (size_t)B_ * N_ * C2_ * sizeof(_Float16); // 4 MB each
  _Float16* Qh = (_Float16*)(ws);
  _Float16* Kh = (_Float16*)(ws + qkBytes);
  _Float16* Vh = (_Float16*)(ws + 2 * qkBytes);                    // 8 MB

  proj_kernel<<<dim3(B_, N_ / 64), 256, 0, stream>>>(x, mask, skin, Wq, bq, Wk, bk, Wv, bv, Qh, Kh, Vh);
  attn_kernel<<<dim3(B_, N_ / 128), 256, 0, stream>>>(Qh, Kh, Vh, x, mask, skin, gamma, out);
}
